// TimeSeriesPredictionModel_82995948028060
// MI455X (gfx1250) — compile-verified, matrix operand order checked by reading, not run on hardware
//
#include <hip/hip_runtime.h>
#include <hip/hip_bf16.h>

typedef __bf16 bf16;
typedef __attribute__((ext_vector_type(16))) __bf16 v16bf;
typedef __attribute__((ext_vector_type(8)))  __bf16 v8bf;
typedef __attribute__((ext_vector_type(8)))  float  v8f;
typedef __attribute__((ext_vector_type(4)))  unsigned v4u;
typedef __attribute__((ext_vector_type(8)))  int      v8i;
typedef __attribute__((ext_vector_type(4)))  int      v4i;
typedef __attribute__((address_space(3)))    v4i      lds_v4i;

#define HSZ    1024            // hidden size
#define GSZ    3072            // 3*H
#define BATCH  128
#define TSTEPS 384
#define MTOT   (TSTEPS * BATCH)
#define LDT    40              // LDS row stride (elements): 64B data + 16B pad = 80B

#define HAS_ASYNC_LDS __has_builtin(__builtin_amdgcn_global_load_async_to_lds_b128)
#define HAS_TDM       __has_builtin(__builtin_amdgcn_tensor_load_to_lds)

// ---------------- device helpers ----------------

__device__ __forceinline__ float sigmoid_f(float x) {
    return 1.0f / (1.0f + __expf(-x));
}
__device__ __forceinline__ float tanh_f(float x) {
    float e = __expf(-2.0f * x);
    return (1.0f - e) / (1.0f + e);
}

// 16-byte global -> LDS copy: async (ASYNCcnt) when available, else VGPR staging.
__device__ __forceinline__ void cp16_g2l(bf16* lds_dst, const bf16* gsrc) {
#if HAS_ASYNC_LDS
    __builtin_amdgcn_global_load_async_to_lds_b128(
        (v4i*)gsrc,
        (lds_v4i*)(unsigned)(size_t)lds_dst,   // low 32 bits of generic LDS ptr = LDS offset
        0, 0);
#else
    *(v8bf*)lds_dst = *(const v8bf*)gsrc;
#endif
}

__device__ __forceinline__ void wait_async() {
#if HAS_ASYNC_LDS
#if __has_builtin(__builtin_amdgcn_s_wait_asynccnt)
    __builtin_amdgcn_s_wait_asynccnt(0);
#else
    asm volatile("s_wait_asynccnt 0x0" ::: "memory");
#endif
#endif
}

#if HAS_TDM
// TDM load of a [planes][rows][cols] bf16 tile (cols contiguous, rows strided by
// row_stride elems, planes strided by plane_stride elems) into LDS at lds_off.
// D# per CDNA5 ISA 8.3/8.4: count=1, type=2 (image), data_size=1 (2 bytes),
// pad_enable with pad_interval=16 DWORDs (code 3) and pad_amount=4 DWORDs
// (code 3): 64B row + 16B pad == LDT*2 LDS row stride.
__device__ __forceinline__ void tdm_load(const void* gsrc, unsigned lds_off,
                                         unsigned cols, unsigned rows, unsigned planes,
                                         unsigned row_stride,
                                         unsigned long long plane_stride) {
    unsigned long long ga = (unsigned long long)(size_t)gsrc;
    v4u g0 = { 1u,                                   // count=1, user descriptor
               lds_off,                              // lds_addr
               (unsigned)ga,                         // global_addr[31:0]
               (unsigned)((ga >> 32) & 0x1FFFFFFu) | (2u << 30) };  // addr[56:32] | type=2
    const unsigned td0 = 1u << 20, td1 = 1u << 20;   // generous tensor dims (no OOB clamp)
    v8i g1;
    g1[0] = (int)((1u << 16) | (1u << 20) | (3u << 22) | (3u << 25));
    //            data_size=1   pad_enable   pad_intvl=3   pad_amount=3
    g1[1] = (int)((td0 & 0xffffu) << 16);                              // tensor_dim0 lo16
    g1[2] = (int)(((td0 >> 16) & 0xffffu) | ((td1 & 0xffffu) << 16));  // td0 hi16 | td1 lo16
    g1[3] = (int)(((td1 >> 16) & 0xffffu) | (cols << 16));             // td1 hi16 | tile_dim0
    g1[4] = (int)(rows | (planes << 16));                              // tile_dim1 | tile_dim2
    g1[5] = (int)row_stride;                                           // dim0_stride[31:0]
    g1[6] = (int)((unsigned)(plane_stride & 0xffffu) << 16);           // dim0_str hi16=0 | dim1_str lo16
    g1[7] = (int)((unsigned)(plane_stride >> 16));                     // dim1_stride[47:16]
    v4i g2 = { 16, 0, 0, 0 };   // tensor_dim2 >= planes
    v4i g3 = { 0, 0, 0, 0 };
    v8i g4 = { 0, 0, 0, 0, 0, 0, 0, 0 };
    __builtin_amdgcn_tensor_load_to_lds(g0, g1, g2, g3, g4, 0);
}
#endif

__device__ __forceinline__ void wait_tensor_tid0(int tid) {
#if HAS_TDM
    if (tid == 0) __builtin_amdgcn_s_wait_tensorcnt(0);
#endif
}

// Load a 16x32 bf16 fragment (A layout; B of W^T is symmetric) from an LDS tile
// with row stride LDT. Per ISA 16-bit layout: lanes 0-15 hold K[0:8],K[16:24];
// lanes 16-31 hold K[8:16],K[24:32]. Two 16-byte ds_load_b128 per lane.
__device__ __forceinline__ v16bf load_frag(const bf16* base) {
    int lane = threadIdx.x & 31;
    int r    = lane & 15;
    int hi   = lane >> 4;
    const bf16* p = base + r * LDT;
    union { v16bf v; v8bf h[2]; } u;
    u.h[0] = *(const v8bf*)(p + (hi ? 8 : 0));
    u.h[1] = *(const v8bf*)(p + (hi ? 24 : 16));
    return u.v;
}

// ---------------- kernels ----------------

__global__ void wcvt(const float* __restrict__ s, bf16* __restrict__ d, int n) {
    int i = blockIdx.x * blockDim.x + threadIdx.x;
    if (i < n) d[i] = (bf16)s[i];
}

// (B,T,H) fp32 -> (T,B,H) bf16
__global__ void conv_seq(const float* __restrict__ in, bf16* __restrict__ out) {
    int row = blockIdx.x;                // row = t*B + b
    int t = row / BATCH, b = row % BATCH;
    const float* src = in + ((size_t)b * TSTEPS + t) * HSZ;
    bf16* dst = out + (size_t)row * HSZ;
    for (int h = threadIdx.x; h < HSZ; h += blockDim.x)
        dst[h] = (bf16)src[h];
}

__global__ void init_h(const float* __restrict__ h0, float* __restrict__ hf,
                       bf16* __restrict__ hb) {
    int i = blockIdx.x * blockDim.x + threadIdx.x;
    float v = h0 ? h0[i] : 0.0f;
    hf[i] = v;
    hb[i] = (bf16)v;
}

// XG[M,3072] = X[M,1024](bf16) @ W[3072,1024]^T (bf16) + bih, fp32 out.
// Block: 256 thr / 8 waves computes 128x128; wave = 16 rows x 128 cols (8 tiles).
// B tile (weights) via TDM; A tile via async-to-LDS.
__global__ void __launch_bounds__(256) proj_gemm(const bf16* __restrict__ X,
                                                 const bf16* __restrict__ W,
                                                 const float* __restrict__ bih,
                                                 float* __restrict__ XG) {
    __shared__ alignas(16) bf16 sA[128 * LDT];
    __shared__ alignas(16) bf16 sB[128 * LDT];
    const int tid = threadIdx.x;
    const int lane = tid & 31, wv = tid >> 5;
    const int n0 = blockIdx.x * 128;
    const int m0 = blockIdx.y * 128;

    v8f acc[8] = {};

    for (int kk = 0; kk < HSZ; kk += 32) {
#if HAS_TDM
        if (tid == 0)
            tdm_load(&W[(size_t)n0 * HSZ + kk], (unsigned)(size_t)&sB[0],
                     32, 128, 1, HSZ, 0ull);
#endif
#pragma unroll
        for (int i = 0; i < 2; ++i) {            // A: 512 16B chunks / 256 threads
            int idx = tid + i * 256;
            int r = idx >> 2, ch = (idx & 3) << 3;
            cp16_g2l(&sA[r * LDT + ch], &X[(size_t)(m0 + r) * HSZ + kk + ch]);
#if !HAS_TDM
            cp16_g2l(&sB[r * LDT + ch], &W[(size_t)(n0 + r) * HSZ + kk + ch]);
#endif
        }
        if (kk + 32 < HSZ)
            __builtin_prefetch(&W[(size_t)(n0 + (tid >> 1)) * HSZ + kk + 32], 0, 1);
        wait_async();
        wait_tensor_tid0(tid);
        __syncthreads();

        v16bf af = load_frag(&sA[wv * 16 * LDT]);
#pragma unroll
        for (int j = 0; j < 8; ++j) {
            v16bf bfrag = load_frag(&sB[j * 16 * LDT]);
            acc[j] = __builtin_amdgcn_wmma_f32_16x16x32_bf16(
                false, af, false, bfrag, (short)0, acc[j], false, false);
        }
        __syncthreads();
    }

    const int rbase = m0 + wv * 16 + ((lane >> 4) << 3);   // C row = a + 8*(lane/16)
#pragma unroll
    for (int j = 0; j < 8; ++j) {
        int col = n0 + j * 16 + (lane & 15);
        float bv = bih[col];
#pragma unroll
        for (int a = 0; a < 8; ++a)
            XG[(size_t)(rbase + a) * GSZ + col] = acc[j][a] + bv;
    }
}

// One GRU time step, fused: gh = h_bf @ Whh^T for gate slices {j0, H+j0, 2H+j0},
// then gates + state update. Grid: 32 blocks (32 cols each), 256 thr / 8 waves.
// All 3 gate weight tiles fetched by ONE 3D TDM descriptor per K chunk.
__global__ void __launch_bounds__(256) gru_step(const float* __restrict__ xg_t,
                                                const float* __restrict__ hpf,
                                                const bf16*  __restrict__ hpb,
                                                const bf16*  __restrict__ Whh,
                                                const float* __restrict__ bhh,
                                                float* __restrict__ hnf,
                                                bf16*  __restrict__ hnb,
                                                bf16*  __restrict__ y_out) {
    __shared__ alignas(16) bf16 sA[128 * LDT];
    __shared__ alignas(16) bf16 sB[3 * 32 * LDT];
    const int tid = threadIdx.x;
    const int lane = tid & 31, wv = tid >> 5;
    const int j0 = blockIdx.x * 32;

    v8f acc[3][2] = {};

    for (int kk = 0; kk < HSZ; kk += 32) {
#if HAS_TDM
        if (tid == 0)   // planes=3 gates, plane stride = H*H elements
            tdm_load(&Whh[(size_t)j0 * HSZ + kk], (unsigned)(size_t)&sB[0],
                     32, 32, 3, HSZ, (unsigned long long)HSZ * HSZ);
#else
        for (int idx = tid; idx < 384; idx += 256) {  // B: 3 x 32 rows x 32 cols
            int g = idx >> 7, rem = idx & 127;
            int r = rem >> 2, ch = (rem & 3) << 3;
            cp16_g2l(&sB[(g * 32 + r) * LDT + ch],
                     &Whh[(size_t)(g * HSZ + j0 + r) * HSZ + kk + ch]);
        }
#endif
#pragma unroll
        for (int i = 0; i < 2; ++i) {            // A: 128 rows x 32 cols
            int idx = tid + i * 256;
            int r = idx >> 2, ch = (idx & 3) << 3;
            cp16_g2l(&sA[r * LDT + ch], &hpb[(size_t)r * HSZ + kk + ch]);
        }
        if (kk + 32 < HSZ)
            __builtin_prefetch(&Whh[(size_t)(j0 + (tid & 31)) * HSZ + kk + 32], 0, 1);
        wait_async();
        wait_tensor_tid0(tid);
        __syncthreads();

        v16bf af = load_frag(&sA[wv * 16 * LDT]);
#pragma unroll
        for (int g = 0; g < 3; ++g) {
#pragma unroll
            for (int c = 0; c < 2; ++c) {
                v16bf bfrag = load_frag(&sB[(g * 32 + c * 16) * LDT]);
                acc[g][c] = __builtin_amdgcn_wmma_f32_16x16x32_bf16(
                    false, af, false, bfrag, (short)0, acc[g][c], false, false);
            }
        }
        __syncthreads();
    }

    const int rbase = wv * 16 + ((lane >> 4) << 3);
#pragma unroll
    for (int c = 0; c < 2; ++c) {
        int col = j0 + c * 16 + (lane & 15);
        float br = bhh[col], bz = bhh[HSZ + col], bn = bhh[2 * HSZ + col];
#pragma unroll
        for (int a = 0; a < 8; ++a) {
            int b = rbase + a;
            const float* xrow = xg_t + (size_t)b * GSZ;
            float r  = sigmoid_f(xrow[col]           + acc[0][c][a] + br);
            float z  = sigmoid_f(xrow[HSZ + col]     + acc[1][c][a] + bz);
            float n  = tanh_f   (xrow[2 * HSZ + col] + r * (acc[2][c][a] + bn));
            float hn = (1.0f - z) * n + z * hpf[(size_t)b * HSZ + col];
            hnf[(size_t)b * HSZ + col] = hn;
            bf16 hb = (bf16)hn;
            hnb[(size_t)b * HSZ + col]   = hb;
            y_out[(size_t)b * HSZ + col] = hb;
        }
    }
}

// predictions[b] = dot(h_last[b,:], lin_W[0,:]) + lin_b
__global__ void linear_k(const float* __restrict__ h, const float* __restrict__ W,
                         const float* __restrict__ bias, float* __restrict__ out) {
    __shared__ float red[8];
    int b = blockIdx.x, tid = threadIdx.x;
    float s = 0.0f;
    for (int i = tid; i < HSZ; i += 256)
        s += h[(size_t)b * HSZ + i] * W[i];
    for (int off = 16; off; off >>= 1)
        s += __shfl_down(s, off, 32);
    if ((tid & 31) == 0) red[tid >> 5] = s;
    __syncthreads();
    if (tid == 0) {
        float t = 0.0f;
        for (int w = 0; w < 8; ++w) t += red[w];
        out[b] = t + bias[0];
    }
}

// ---------------- host ----------------

extern "C" void kernel_launch(void* const* d_in, const int* in_sizes, int n_in,
                              void* d_out, int out_size, void* d_ws, size_t ws_size,
                              hipStream_t stream) {
    const float* rand_input = (const float*)d_in[1];
    const float* h0    = (const float*)d_in[2];
    const float* Wih0  = (const float*)d_in[7];
    const float* Whh0  = (const float*)d_in[8];
    const float* bih0  = (const float*)d_in[9];
    const float* bhh0  = (const float*)d_in[10];
    const float* Wih1  = (const float*)d_in[11];
    const float* Whh1  = (const float*)d_in[12];
    const float* bih1  = (const float*)d_in[13];
    const float* bhh1  = (const float*)d_in[14];
    const float* linW  = (const float*)d_in[15];
    const float* linb  = (const float*)d_in[16];
    float* out = (float*)d_out;

    char* w = (char*)d_ws;
    size_t off = 0;
    auto alloc = [&](size_t bytes) {
        void* p = w + off;
        off = (off + bytes + 255) & ~(size_t)255;
        return p;
    };
    bf16*  Wih0b = (bf16*)alloc((size_t)GSZ * HSZ * 2);
    bf16*  Whh0b = (bf16*)alloc((size_t)GSZ * HSZ * 2);
    bf16*  Wih1b = (bf16*)alloc((size_t)GSZ * HSZ * 2);
    bf16*  Whh1b = (bf16*)alloc((size_t)GSZ * HSZ * 2);
    bf16*  seqA  = (bf16*)alloc((size_t)MTOT * HSZ * 2);
    bf16*  seqB  = (bf16*)alloc((size_t)MTOT * HSZ * 2);
    float* xg    = (float*)alloc((size_t)MTOT * GSZ * 4);
    float* hF0   = (float*)alloc((size_t)BATCH * HSZ * 4);
    float* hF1   = (float*)alloc((size_t)BATCH * HSZ * 4);
    bf16*  hB0   = (bf16*)alloc((size_t)BATCH * HSZ * 2);
    bf16*  hB1   = (bf16*)alloc((size_t)BATCH * HSZ * 2);
    float* hF[2] = { hF0, hF1 };
    bf16*  hB[2] = { hB0, hB1 };

    const int wn = GSZ * HSZ;
    wcvt<<<(wn + 255) / 256, 256, 0, stream>>>(Wih0, Wih0b, wn);
    wcvt<<<(wn + 255) / 256, 256, 0, stream>>>(Whh0, Whh0b, wn);
    wcvt<<<(wn + 255) / 256, 256, 0, stream>>>(Wih1, Wih1b, wn);
    wcvt<<<(wn + 255) / 256, 256, 0, stream>>>(Whh1, Whh1b, wn);
    conv_seq<<<MTOT, 256, 0, stream>>>(rand_input, seqA);

    auto run_layer = [&](const bf16* X, const bf16* Wihb, const bf16* Whhb,
                         const float* bih, const float* bhh,
                         const float* h0p, bf16* Y) {
        init_h<<<(BATCH * HSZ) / 256, 256, 0, stream>>>(h0p, hF[0], hB[0]);
        proj_gemm<<<dim3(GSZ / 128, MTOT / 128), 256, 0, stream>>>(X, Wihb, bih, xg);
        for (int t = 0; t < TSTEPS; ++t) {
            int cur = t & 1, nxt = cur ^ 1;
            gru_step<<<HSZ / 32, 256, 0, stream>>>(
                xg + (size_t)t * BATCH * GSZ, hF[cur], hB[cur], Whhb, bhh,
                hF[nxt], hB[nxt], Y + (size_t)t * BATCH * HSZ);
        }
    };

    // Run 1 (h0 initial states)
    run_layer(seqA, Wih0b, Whh0b, bih0, bhh0, h0,                       seqB);
    run_layer(seqB, Wih1b, Whh1b, bih1, bhh1, h0 + (size_t)BATCH * HSZ, seqA);
    // split/concat along T is identity -> Run 2 (zero initial states)
    run_layer(seqA, Wih0b, Whh0b, bih0, bhh0, nullptr, seqB);
    run_layer(seqB, Wih1b, Whh1b, bih1, bhh1, nullptr, seqA);

    // T=384 is even: final hidden of last pass lands in hF[0]
    linear_k<<<BATCH, 256, 0, stream>>>(hF[0], linW, linb, out);
}
